// GCNNet_40570261078534
// MI455X (gfx1250) — compile-verified, matrix operand order checked by reading, not run on hardware
//
#include <hip/hip_runtime.h>
#include <math.h>

typedef __attribute__((ext_vector_type(16))) _Float16 v16h;
typedef __attribute__((ext_vector_type(8)))  _Float16 v8h;
typedef __attribute__((ext_vector_type(8)))  float    v8f;

// ---------------------------------------------------------------------------
// Degree / symmetric normalization: deg[i] = 1 (self loop) + in-degree,
// then dis[i] = rsqrt(max(deg,1)).  Computed in place in the dis buffer.
// ---------------------------------------------------------------------------
__global__ void k_init_deg(float* __restrict__ deg, int N) {
  int i = blockIdx.x * blockDim.x + threadIdx.x;
  if (i < N) deg[i] = 1.0f;
}

__global__ void k_deg_accum(const int* __restrict__ dst, float* __restrict__ deg, int E) {
  int i = blockIdx.x * blockDim.x + threadIdx.x;
  if (i < E) atomicAdd(&deg[dst[i]], 1.0f);
}

__global__ void k_deg_to_dis(float* __restrict__ deg, int N) {
  int i = blockIdx.x * blockDim.x + threadIdx.x;
  if (i < N) deg[i] = rsqrtf(fmaxf(deg[i], 1.0f));
}

__global__ void k_zero4(float4* __restrict__ p, long long n4) {
  long long i = (long long)blockIdx.x * blockDim.x + threadIdx.x;
  if (i < n4) p[i] = make_float4(0.f, 0.f, 0.f, 0.f);
}

// ---------------------------------------------------------------------------
// WMMA GEMM:  O[Nrows, NCOL] = X[Nrows, 128] @ W[128, NCOL]   (row-major)
// NCOL is compile-time (128 or 40).  One wave per 16x16 tile, 8 waves/block.
// W is pre-swizzled into LDS in exact B-fragment order: for each
// (k-chunk, col-tile, lane) a contiguous run of 16 f16 (stride padded to 24
// f16 = 48B to avoid the stride-32B LDS bank-conflict pattern), so each wave
// reads its B operand as two ds_load_b128.
// Lane layouts per CDNA5 ISA §7.12.2 (wave32).
// ---------------------------------------------------------------------------
template <int NCOL>
__global__ __launch_bounds__(256) void k_gemm_wmma(
    const float* __restrict__ X, const float* __restrict__ W,
    float* __restrict__ O, int Nrows)
{
  constexpr int NT = (NCOL + 15) / 16;        // col tiles
  constexpr int FRAG = 24;                    // padded f16 stride per lane frag
  __shared__ _Float16 Wl[4 * NT * 32 * FRAG]; // 4 K-chunks of fragments

  // cooperative fill: j -> (kcI, tile, lane, i) fragment element
  constexpr int TOT = 4 * NT * 32 * 16;
  for (int j = threadIdx.x; j < TOT; j += blockDim.x) {
    const int i    = j & 15;
    const int lane = (j >> 4) & 31;
    const int rest = j >> 9;                  // kcI*NT + tile
    const int tile = rest % NT;
    const int kcI  = rest / NT;
    const int k    = kcI * 32 + ((lane >> 4) << 3) + ((i & 8) << 1) + (i & 7);
    int col = tile * 16 + (lane & 15);
    if (col >= NCOL) col = NCOL - 1;          // clamped; stores are guarded
    Wl[(((kcI * NT + tile) << 5) + lane) * FRAG + i] = (_Float16)W[k * NCOL + col];
  }
  __syncthreads();

  const int lane = threadIdx.x & 31;
  const int wv   = threadIdx.x >> 5;
  if (wv >= NT) return;                       // idle waves (NCOL=40 case)

  const int m0   = blockIdx.x * 16;
  const int t    = lane & 15;                 // A-row / B,C-col within tile
  const int half = lane >> 4;
  const int col  = wv * 16 + t;

  int arow = m0 + t; if (arow >= Nrows) arow = Nrows - 1;
  const float* xrow = X + (size_t)arow * 128;

  v8f acc = {};
#pragma unroll
  for (int kc = 0; kc < 128; kc += 32) {
    // A 16x32 f16: half 0 -> K = kc+[0..7], kc+16+[0..7]; half 1 -> +8
    v16h a;
    const float* ap = xrow + kc + half * 8;
#pragma unroll
    for (int i = 0; i < 8; ++i) {
      a[i]     = (_Float16)ap[i];
      a[8 + i] = (_Float16)ap[16 + i];
    }
    // B fragment: two contiguous 16B LDS loads
    const v8h* bp = (const v8h*)&Wl[((((kc >> 5) * NT + wv) << 5) + lane) * FRAG];
    const v8h blo = bp[0];
    const v8h bhi = bp[1];
    const v16h b = __builtin_shufflevector(blo, bhi,
        0, 1, 2, 3, 4, 5, 6, 7, 8, 9, 10, 11, 12, 13, 14, 15);

    acc = __builtin_amdgcn_wmma_f32_16x16x32_f16(
        /*neg_a=*/false, a, /*neg_b=*/false, b,
        /*c_mod=*/(short)0, acc, /*reuse_a=*/false, /*reuse_b=*/false);
  }

  // C/D f32 16x16: VGPR r -> M = r + 8*half, N = t
  float* op = O + (size_t)(m0 + 8 * half) * NCOL + col;
  const bool fast = (m0 + 16 <= Nrows) && ((NCOL % 16 == 0) || (wv != NT - 1));
  if (fast) {
#pragma unroll
    for (int r = 0; r < 8; ++r) op[(size_t)r * NCOL] = acc[r];
  } else {
#pragma unroll
    for (int r = 0; r < 8; ++r)
      if ((m0 + 8 * half + r) < Nrows && col < NCOL) op[(size_t)r * NCOL] = acc[r];
  }
}

// ---------------------------------------------------------------------------
// Edge aggregation: out[dst] += H[src] * dis[src]*dis[dst]; self loops are
// virtual edges e in [E, E+N).  One wave per edge; d=128 -> float4 per lane.
// ---------------------------------------------------------------------------
__global__ __launch_bounds__(256) void k_aggregate128(
    const float* __restrict__ H, const int* __restrict__ src,
    const int* __restrict__ dst, const float* __restrict__ dis,
    float* __restrict__ out, int E, int N)
{
  const int lane = threadIdx.x & 31;
  const long long e = (long long)blockIdx.x * 8 + (threadIdx.x >> 5);
  if (e >= (long long)E + N) return;
  int s, d; float w;
  if (e < E) { s = src[e]; d = dst[e]; w = dis[s] * dis[d]; }
  else       { s = d = (int)(e - E); float ds_ = dis[s]; w = ds_ * ds_; }

  const float4 v = *(const float4*)(H + (size_t)s * 128 + lane * 4);
  float* o = out + (size_t)d * 128 + lane * 4;
  atomicAdd(o + 0, v.x * w);
  atomicAdd(o + 1, v.y * w);
  atomicAdd(o + 2, v.z * w);
  atomicAdd(o + 3, v.w * w);
}

__global__ __launch_bounds__(256) void k_aggregate40(
    const float* __restrict__ H, const int* __restrict__ src,
    const int* __restrict__ dst, const float* __restrict__ dis,
    float* __restrict__ out, int E, int N)
{
  const int lane = threadIdx.x & 31;
  const long long e = (long long)blockIdx.x * 8 + (threadIdx.x >> 5);
  if (e >= (long long)E + N) return;
  int s, d; float w;
  if (e < E) { s = src[e]; d = dst[e]; w = dis[s] * dis[d]; }
  else       { s = d = (int)(e - E); float ds_ = dis[s]; w = ds_ * ds_; }

  const float* hp = H + (size_t)s * 40;
  float* o = out + (size_t)d * 40;
  for (int c = lane; c < 40; c += 32)
    atomicAdd(o + c, hp[c] * w);
}

__global__ void k_bias_relu(float* __restrict__ h, const float* __restrict__ b,
                            long long total) {
  long long i = (long long)blockIdx.x * blockDim.x + threadIdx.x;
  if (i < total) {
    float v = h[i] + b[(int)(i & 127)];
    h[i] = v > 0.0f ? v : 0.0f;
  }
}

// ---------------------------------------------------------------------------
// In-place log_softmax over 40 classes (+ bias b2), one wave32 per row.
// ---------------------------------------------------------------------------
__global__ __launch_bounds__(256) void k_logsoftmax40(
    float* __restrict__ out, const float* __restrict__ b2, int N)
{
  const int lane = threadIdx.x & 31;
  const int row  = blockIdx.x * 8 + (threadIdx.x >> 5);
  if (row >= N) return;
  float* p = out + (size_t)row * 40;

  float v0 = p[lane] + b2[lane];
  float v1 = (lane < 8) ? (p[32 + lane] + b2[32 + lane]) : -3.402823466e38f;

  float m = fmaxf(v0, v1);
  for (int off = 16; off > 0; off >>= 1) m = fmaxf(m, __shfl_xor(m, off, 32));

  float s = expf(v0 - m) + ((lane < 8) ? expf(v1 - m) : 0.0f);
  for (int off = 16; off > 0; off >>= 1) s += __shfl_xor(s, off, 32);

  float ls = logf(s);
  p[lane] = v0 - m - ls;
  if (lane < 8) p[32 + lane] = v1 - m - ls;
}

// ---------------------------------------------------------------------------
extern "C" void kernel_launch(void* const* d_in, const int* in_sizes, int n_in,
                              void* d_out, int out_size, void* d_ws, size_t ws_size,
                              hipStream_t stream) {
  (void)n_in; (void)out_size; (void)ws_size;
  const float* x  = (const float*)d_in[0];
  const int*   ei = (const int*)d_in[1];     // edge_index, [2, E] row-major
  const float* W1 = (const float*)d_in[2];
  const float* b1 = (const float*)d_in[3];
  const float* Wi = (const float*)d_in[4];
  const float* bi = (const float*)d_in[5];
  const float* W2 = (const float*)d_in[6];
  const float* b2 = (const float*)d_in[7];

  const int N = in_sizes[0] / 128;
  const int E = in_sizes[1] / 2;
  const int C = 40;
  const int* srcA = ei;
  const int* dstA = ei + E;

  float* dis = (float*)d_ws;                 // N
  float* F   = dis + N;                      // N*128 feature buffer
  float* G   = F + (size_t)N * 128;          // N*128 gemm-out buffer
  float* out = (float*)d_out;                // N*40

  const int TB = 256;
  const long long NH = (long long)N * 128;
  const long long NC = (long long)N * C;
  const int aggBlocks = (int)(((long long)E + N + 7) / 8);
  const int rowTiles  = (N + 15) / 16;
  const int nBlk   = (N + TB - 1) / TB;
  const int eBlk   = (E + TB - 1) / TB;
  const int nhBlk  = (int)((NH + TB - 1) / TB);
  const int nh4Blk = (int)((NH / 4 + TB - 1) / TB);
  const int nc4Blk = (int)((NC / 4 + TB - 1) / TB);

  // normalization:  dis = rsqrt(deg),  deg = 1 + in-degree
  k_init_deg  <<<nBlk, TB, 0, stream>>>(dis, N);
  k_deg_accum <<<eBlk, TB, 0, stream>>>(dstA, dis, E);
  k_deg_to_dis<<<nBlk, TB, 0, stream>>>(dis, N);

  // layer 1: G = x@W1 ; F = scatter(G) ; F = relu(F + b1)
  k_gemm_wmma<128><<<rowTiles, TB, 0, stream>>>(x, W1, G, N);
  k_zero4       <<<nh4Blk, TB, 0, stream>>>((float4*)F, NH / 4);
  k_aggregate128<<<aggBlocks, TB, 0, stream>>>(G, srcA, dstA, dis, F, E, N);
  k_bias_relu   <<<nhBlk, TB, 0, stream>>>(F, b1, NH);

  // layer 2 (inner): G = F@Wi ; F = scatter(G) ; F = relu(F + bi)
  k_gemm_wmma<128><<<rowTiles, TB, 0, stream>>>(F, Wi, G, N);
  k_zero4       <<<nh4Blk, TB, 0, stream>>>((float4*)F, NH / 4);
  k_aggregate128<<<aggBlocks, TB, 0, stream>>>(G, srcA, dstA, dis, F, E, N);
  k_bias_relu   <<<nhBlk, TB, 0, stream>>>(F, bi, NH);

  // layer 3: G[:, :40] = F@W2 ; out = scatter(G) ; out = log_softmax(out + b2)
  k_gemm_wmma<40><<<rowTiles, TB, 0, stream>>>(F, W2, G, N);
  k_zero4       <<<nc4Blk, TB, 0, stream>>>((float4*)out, NC / 4);
  k_aggregate40 <<<aggBlocks, TB, 0, stream>>>(G, srcA, dstA, dis, out, E, N);
  k_logsoftmax40<<<(N + 7) / 8, TB, 0, stream>>>(out, b2, N);
}
